// AttentionModule_87694642250031
// MI455X (gfx1250) — matrix operand order repaired and checked by reading, not compiled, not run on hardware
//
#include <hip/hip_runtime.h>
#include <hip/hip_bf16.h>
#include <math.h>

typedef _Float16 v8h  __attribute__((ext_vector_type(8)));
typedef _Float16 v16h __attribute__((ext_vector_type(16)));
typedef float    v8f  __attribute__((ext_vector_type(8)));

#define DD 128

// ---------------------------------------------------------------------------
// WMMA helpers (CDNA5 v_wmma_f32_16x16x32_f16, wave32)
// ---------------------------------------------------------------------------
__device__ __forceinline__ v8f wmma_f16(v16h a, v16h b, v8f c) {
    // 8 args: (neg_a, A, neg_b, B, c_mod, C, reuse_a, reuse_b)
    return __builtin_amdgcn_wmma_f32_16x16x32_f16(false, a, false, b,
                                                  (short)0, c, false, false);
}

// A fragment: 16x32 f16 tile, row-major in LDS with row stride `ld` halves.
// Lane L holds row M = L&15; half h holds K = (h/8)*16 + (L/16)*8 + (h%8).
__device__ __forceinline__ v16h load_a_frag(const _Float16* s, int ld, int k0, int lane) {
    const int row = lane & 15;
    const int hi8 = (lane >> 4) << 3;
    const _Float16* p = s + row * ld + k0 + hi8;
    v8h lo = *(const v8h*)(p);
    v8h hh = *(const v8h*)(p + 16);
    return __builtin_shufflevector(lo, hh, 0,1,2,3,4,5,6,7,8,9,10,11,12,13,14,15);
}

// B fragment: 32x16 (KxN) with B stored transposed in LDS as [N][K] (stride ld).
// Lane L holds col N = L&15; half h holds K = k0 + (h/8)*16 + (L/16)*8 + (h%8).
__device__ __forceinline__ v16h load_b_frag(const _Float16* sT, int ld, int n0, int k0, int lane) {
    const int n  = lane & 15;
    const int hi = lane >> 4;
    const _Float16* p = sT + (size_t)(n0 + n) * ld + k0 + 8 * hi;
    v8h lo = *(const v8h*)(p);
    v8h hh = *(const v8h*)(p + 16);
    return __builtin_shufflevector(lo, hh, 0,1,2,3,4,5,6,7,8,9,10,11,12,13,14,15);
}

// ---------------------------------------------------------------------------
// Branch-free fast transcendentals (v_exp_f32 + v_rcp_f32, no libm expansion)
//   tanh(v)   = 1 - 2/(exp(2v)+1)      (exp->inf => 1; exp->0 => -1; no NaN)
//   sigmoid(v)= 1/(1+exp(-v))
// ---------------------------------------------------------------------------
__device__ __forceinline__ float fast_tanh(float v) {
    float e = __expf(2.0f * v);
    return 1.0f - 2.0f * __builtin_amdgcn_rcpf(e + 1.0f);
}
__device__ __forceinline__ float fast_sigmoid(float v) {
    return __builtin_amdgcn_rcpf(1.0f + __expf(-v));
}

// ---------------------------------------------------------------------------
// zero workspace + output
// ---------------------------------------------------------------------------
__global__ void zero_kernel(float* __restrict__ sums, float* __restrict__ cnt,
                            float* __restrict__ outp, int Bseg) {
    int i = blockIdx.x * blockDim.x + threadIdx.x;
    int tot = Bseg * DD;
    if (i < tot) { sums[i] = 0.0f; outp[i] = 0.0f; }
    if (i < Bseg) cnt[i] = 0.0f;
}

// ---------------------------------------------------------------------------
// Fused per-row pipeline. One wave = one 16-row tile; block = 8 waves = 128 rows.
// PHASE 0: x2 segment-sum + count.   PHASE 1: coef = sigmoid(<x2, tg[b]>), out += coef*x2.
// ---------------------------------------------------------------------------
template <int PHASE>
__global__ void __launch_bounds__(256)
fused_rows_kernel(const float* __restrict__ x, const long long* __restrict__ batch,
                  const float* __restrict__ fc1w, const float* __restrict__ fc1b,
                  const float* __restrict__ fc2w, const float* __restrict__ fc2b,
                  const float* __restrict__ tg,
                  float* __restrict__ sums, float* __restrict__ cnt,
                  float* __restrict__ outp, int N) {
    __shared__ __align__(32) _Float16 s_fc1[32 * DD];   // fc1_w as [n=32][k=128]
    __shared__ __align__(32) _Float16 s_fc2[DD * 32];   // fc2_w as [n=128][k=32]
    __shared__ float s_b1[32];
    __shared__ float s_b2[DD];
    __shared__ __align__(32) _Float16 s_x[8][16 * DD];  // per-wave x tile f16 (reused as att in PHASE 1)
    __shared__ __align__(32) _Float16 s_a[8][16 * 32];  // per-wave hidden tile f16

    const int tid  = threadIdx.x;
    const int lane = tid & 31;
    const int w    = tid >> 5;
    const int hi   = lane >> 4;
    const int nn   = lane & 15;

    for (int i = tid; i < 32 * DD; i += 256) s_fc1[i] = (_Float16)fc1w[i];
    for (int i = tid; i < DD * 32; i += 256) s_fc2[i] = (_Float16)fc2w[i];
    if (tid < 32) s_b1[tid] = fc1b[tid];
    if (tid < DD) s_b2[tid] = fc2b[tid];
    __syncthreads();

    const long m0 = (long)blockIdx.x * 128 + (long)w * 16;
    if (m0 >= (long)N) return;

    _Float16* sx = s_x[w];
    _Float16* sa = s_a[w];

    // Stage x tile [16][128] as f16 (coalesced)
    for (int i = lane; i < 16 * DD; i += 32) {
        long row = m0 + (i >> 7);
        float v = (row < (long)N) ? x[row * DD + (i & (DD - 1))] : 0.0f;
        sx[i] = (_Float16)v;
    }

    // Segment ids (batch is sorted -> tile uniform iff endpoints match)
    int segid[8];
#pragma unroll
    for (int r = 0; r < 8; ++r) {
        long row = m0 + r + 8 * hi;
        segid[r] = (int)batch[row < (long)N ? row : (long)N - 1];
    }
    const int b0 = (int)batch[m0];
    const long mlast = (m0 + 15 < (long)N) ? m0 + 15 : (long)N - 1;
    const bool uniform = (b0 == (int)batch[mlast]);

    // ---- GEMM1: h = relu(x @ fc1_w^T + b1)  [16x128]x[128x32] -> s_a [16][32]
#pragma unroll
    for (int nt = 0; nt < 2; ++nt) {
        v8f acc = {0.f, 0.f, 0.f, 0.f, 0.f, 0.f, 0.f, 0.f};
#pragma unroll
        for (int kt = 0; kt < 4; ++kt) {
            v16h a = load_a_frag(sx, DD, kt * 32, lane);
            v16h b = load_b_frag(s_fc1, DD, nt * 16, kt * 32, lane);
            acc = wmma_f16(a, b, acc);
        }
        const int n = nt * 16 + nn;
        const float bia = s_b1[n];
#pragma unroll
        for (int r = 0; r < 8; ++r) {
            float v = acc[r] + bia;
            v = v > 0.0f ? v : 0.0f;
            sa[(r + 8 * hi) * 32 + n] = (_Float16)v;
        }
    }

    // ---- GEMM2: att = tanh(h @ fc2_w^T + b2), x2 = x*(1+att), accumulate
    const v16h a2 = load_a_frag(sa, 32, 0, lane);  // K=32, invariant over nt
    float dpart[8] = {0.f, 0.f, 0.f, 0.f, 0.f, 0.f, 0.f, 0.f};

#pragma unroll
    for (int nt = 0; nt < 8; ++nt) {
        v8f acc = {0.f, 0.f, 0.f, 0.f, 0.f, 0.f, 0.f, 0.f};
        v16h b = load_b_frag(s_fc2, 32, nt * 16, 0, lane);
        acc = wmma_f16(a2, b, acc);

        const int col = nt * 16 + nn;
        const float bia = s_b2[col];
        float tgu = 0.0f;
        if (PHASE == 1 && uniform) tgu = tg[(size_t)b0 * DD + col];
        float colsum = 0.0f;
#pragma unroll
        for (int r = 0; r < 8; ++r) {
            const int M = r + 8 * hi;
            const long row = m0 + M;
            const bool ok = row < (long)N;
            const float att = fast_tanh(acc[r] + bia);
            const float xv = ok ? x[row * DD + col] : 0.0f;
            const float x2 = xv + att * xv;
            if (PHASE == 0) {
                if (uniform) colsum += x2;
                else if (ok) atomicAdd(&sums[(size_t)segid[r] * DD + col], x2);
            } else {
                sx[M * DD + col] = (_Float16)att;  // reuse x-tile buffer as att cache
                const float tv = uniform ? tgu
                                         : (ok ? tg[(size_t)segid[r] * DD + col] : 0.0f);
                dpart[r] += x2 * tv;
            }
        }
        if (PHASE == 0 && uniform) {
            colsum += __shfl_xor(colsum, 16, 32);
            if (hi == 0) atomicAdd(&sums[(size_t)b0 * DD + col], colsum);
        }
    }

    if (PHASE == 0) {
        if (uniform) {
            if (lane == 0) {
                long nrows = (long)N - m0; if (nrows > 16) nrows = 16;
                atomicAdd(&cnt[b0], (float)nrows);
            }
        } else if (lane < 16) {
            long row = m0 + lane;
            if (row < (long)N) atomicAdd(&cnt[(int)batch[row]], 1.0f);
        }
        return;
    }

    // ---- PHASE 1: finish per-row dots -> coef, then weighted scatter
#pragma unroll
    for (int r = 0; r < 8; ++r) {
#pragma unroll
        for (int off = 1; off < 16; off <<= 1)
            dpart[r] += __shfl_xor(dpart[r], off, 32);
    }
    float coef[8];
#pragma unroll
    for (int r = 0; r < 8; ++r) coef[r] = fast_sigmoid(dpart[r]);

#pragma unroll
    for (int nt = 0; nt < 8; ++nt) {
        const int col = nt * 16 + nn;
        float colsum = 0.0f;
#pragma unroll
        for (int r = 0; r < 8; ++r) {
            const int M = r + 8 * hi;
            const long row = m0 + M;
            const bool ok = row < (long)N;
            const float att = (float)sx[M * DD + col];
            const float xv = ok ? x[row * DD + col] : 0.0f;
            const float x2 = xv + att * xv;
            const float wv = coef[r] * x2;
            if (uniform) colsum += wv;
            else if (ok) atomicAdd(&outp[(size_t)segid[r] * DD + col], wv);
        }
        if (uniform) {
            colsum += __shfl_xor(colsum, 16, 32);
            if (hi == 0) atomicAdd(&outp[(size_t)b0 * DD + col], colsum);
        }
    }
}

// ---------------------------------------------------------------------------
// tg = tanh((sums/max(cnt,1)) @ Wm)   [B x 128] x [128 x 128], WMMA
// block = 128 threads = 4 waves = 64 rows
// ---------------------------------------------------------------------------
__global__ void __launch_bounds__(128)
graph_mean_kernel(const float* __restrict__ sums, const float* __restrict__ cnt,
                  const float* __restrict__ Wm, float* __restrict__ tg, int Bseg) {
    __shared__ __align__(32) _Float16 s_wmT[DD * DD];     // Wm^T as [n][k]
    __shared__ __align__(32) _Float16 s_mean[4][16 * DD]; // per-wave mean tile f16

    const int tid  = threadIdx.x;
    const int lane = tid & 31;
    const int w    = tid >> 5;
    const int hi   = lane >> 4;
    const int nn   = lane & 15;

    for (int i = tid; i < DD * DD; i += 128) {
        int k = i >> 7, n = i & (DD - 1);
        s_wmT[n * DD + k] = (_Float16)Wm[i];
    }
    __syncthreads();

    const int m0 = blockIdx.x * 64 + w * 16;
    if (m0 >= Bseg) return;

    _Float16* sm = s_mean[w];
    for (int i = lane; i < 16 * DD; i += 32) {
        int row = m0 + (i >> 7);
        float v = 0.0f;
        if (row < Bseg) {
            float c = cnt[row];
            v = sums[(size_t)row * DD + (i & (DD - 1))] / fmaxf(c, 1.0f);
        }
        sm[i] = (_Float16)v;
    }

#pragma unroll
    for (int nt = 0; nt < 8; ++nt) {
        v8f acc = {0.f, 0.f, 0.f, 0.f, 0.f, 0.f, 0.f, 0.f};
#pragma unroll
        for (int kt = 0; kt < 4; ++kt) {
            v16h a = load_a_frag(sm, DD, kt * 32, lane);
            v16h b = load_b_frag(s_wmT, DD, nt * 16, kt * 32, lane);
            acc = wmma_f16(a, b, acc);
        }
        const int col = nt * 16 + nn;
#pragma unroll
        for (int r = 0; r < 8; ++r) {
            const int row = m0 + r + 8 * hi;
            if (row < Bseg) tg[(size_t)row * DD + col] = fast_tanh(acc[r]);
        }
    }
}

// ---------------------------------------------------------------------------
extern "C" void kernel_launch(void* const* d_in, const int* in_sizes, int n_in,
                              void* d_out, int out_size, void* d_ws, size_t ws_size,
                              hipStream_t stream) {
    const float*     x     = (const float*)d_in[0];
    const long long* batch = (const long long*)d_in[1];
    // d_in[2] = size scalar (B), derived from out_size instead
    const float* Wm   = (const float*)d_in[3];
    const float* fc1w = (const float*)d_in[4];
    const float* fc1b = (const float*)d_in[5];
    const float* fc2w = (const float*)d_in[6];
    const float* fc2b = (const float*)d_in[7];
    float* outp = (float*)d_out;

    const int N    = in_sizes[0] / DD;
    const int Bseg = out_size / DD;

    float* ws   = (float*)d_ws;
    float* sums = ws;                                 // [Bseg*128]
    float* cnt  = sums + (size_t)Bseg * DD;           // [Bseg]
    float* tg   = cnt + Bseg;                         // [Bseg*128]

    const int tot = Bseg * DD;
    zero_kernel<<<(tot + 255) / 256, 256, 0, stream>>>(sums, cnt, outp, Bseg);

    const int gridA = (N + 127) / 128;
    fused_rows_kernel<0><<<gridA, 256, 0, stream>>>(x, batch, fc1w, fc1b, fc2w, fc2b,
                                                    nullptr, sums, cnt, nullptr, N);

    graph_mean_kernel<<<(Bseg + 63) / 64, 128, 0, stream>>>(sums, cnt, Wm, tg, Bseg);

    fused_rows_kernel<1><<<gridA, 256, 0, stream>>>(x, batch, fc1w, fc1b, fc2w, fc2b,
                                                    tg, nullptr, nullptr, outp, N);
}
